// CrossAttention_83872121356514
// MI455X (gfx1250) — matrix operand order repaired and checked by reading, not compiled, not run on hardware
//
#include <hip/hip_runtime.h>
#include <stdint.h>

// ---------------------------------------------------------------------------
// CDNA5 / gfx1250 cross-attention, bf16 WMMA everywhere:
//   Q  = Xq*Wq+bq            (fp32 in -> bf16 out, row-major)
//   K  = Xk*Wk+bk            (fp32 in -> bf16 out, row-major)
//   Vt = (Xk*Wv+bv)^T        (fp32 in -> bf16 out, TRANSPOSED [H,Skv]/batch)
//   S  = (1/32) Q*K^T        (bf16 WMMA, A: async-LDS, B: TDM tensor load)
//   P  = softmax(S+maskbias) (in-place bf16 over fp32 rows)
//   O  = P*Vt^T              (same WMMA kernel)
// Data movement in the attention GEMMs uses BOTH CDNA5 DMA paths:
//   - A tile: per-lane global_load_async_to_lds_b128 (ASYNCcnt)
//   - B tile: one TENSOR_LOAD_TO_LDS descriptor per k-step (TENSORcnt),
//     with TDM hardware padding matching the LDS pitch (64B row + 16B pad).
// ---------------------------------------------------------------------------

typedef __bf16 bf16_t;
typedef __bf16 v16bf __attribute__((ext_vector_type(16)));
typedef __bf16 v8bf  __attribute__((ext_vector_type(8)));
typedef __bf16 v4bf  __attribute__((ext_vector_type(4)));
typedef float  v8f   __attribute__((ext_vector_type(8)));
typedef unsigned int v4u __attribute__((ext_vector_type(4)));
typedef int v8i __attribute__((ext_vector_type(8)));
typedef int v4i __attribute__((ext_vector_type(4)));

constexpr int BM = 128, BN = 128, BK = 32, LK = BK + 8;  // 80B LDS row pitch

// One shared struct => LDS byte offsets are compile-time constants from base 0
// (kernel LDS allocation starts at offset 0; required for asm/TDM LDS addrs).
struct SharedTiles {
  bf16_t As[2][BM][LK];  // [buf][m][k]
  bf16_t Bs[2][BN][LK];  // [buf][n][k]  (K-contiguous rows)
};
constexpr uint32_t LDS_B_BASE = 2u * BM * LK * 2u;  // byte offset of Bs

__device__ __forceinline__ uint32_t lds_off_A(int buf, int r, int c) {
  return (uint32_t)((((buf * BM) + r) * LK + c) * 2);
}
__device__ __forceinline__ uint32_t lds_off_B(int buf, int n, int c) {
  return LDS_B_BASE + (uint32_t)((((buf * BN) + n) * LK + c) * 2);
}

// Async DMA: 16 bytes global -> LDS per lane. Tracked by ASYNCcnt.
__device__ __forceinline__ void async_copy_b128(uint32_t lds_byte_off,
                                                uint64_t gbase,
                                                uint32_t gbyte_off) {
  asm volatile("global_load_async_to_lds_b128 %0, %1, %2"
               :
               : "v"(lds_byte_off), "v"(gbyte_off), "s"(gbase)
               : "memory");
}
__device__ __forceinline__ void wait_async0() {
  asm volatile("s_wait_asynccnt 0x0" ::: "memory");
}

// ---------------------------------------------------------------------------
// TDM: load a 2D tile [rows x 32 bf16] (row stride = stride_elems, bf16) from
// global into LDS at lds_byte, with LDS padding: 64B of data then 16B pad per
// row => matches the LK=40 bf16 pitch. D# packed per ISA ch.8:
//  g0: count=1 | lds_addr | global_addr[56:0] | type=2
//  g1: data_size=1(2B), pad_enable, pad_interval=3(64B), pad_amount=3(4 dw),
//      tensor_dim0/1, tile_dim0=32, tile_dim1=rows, tensor_dim0_stride
//  g2/g3 (+extra group on this toolchain): zero (2D tile).
// Tracked by TENSORcnt; EXEC ignored (issued per-wave, wave 0 only).
// This toolchain's builtin is the 6-arg form:
//   (uint32x4 g0, int32x8 g1, int32x4, int32x4, int32x8, i32 cpol)
// ---------------------------------------------------------------------------
__device__ __forceinline__ void tdm_load_tile_2d(uint32_t lds_byte,
                                                 uint64_t gaddr, uint32_t rows,
                                                 uint32_t stride_elems) {
  const uint32_t tdim0 = stride_elems;  // >= tile_dim0: no OOB clipping
  const uint32_t tdim1 = rows;
  v4u g0;
  g0.x = 1u;                                      // count=1 (valid user D#)
  g0.y = lds_byte;                                // lds_addr
  g0.z = (uint32_t)gaddr;                         // global_addr[31:0]
  g0.w = (uint32_t)(gaddr >> 32) | (2u << 30);    // global_addr[56:32]|type=2
  v8i g1;
  g1[0] = (1 << 16) | (1 << 20) | (3 << 22) | (3 << 25);
  g1[1] = (int)((tdim0 & 0xffffu) << 16);                    // tensor_dim0 lo
  g1[2] = (int)((tdim0 >> 16) | ((tdim1 & 0xffffu) << 16));  // d0 hi, d1 lo
  g1[3] = (int)((tdim1 >> 16) | (32u << 16));                // d1 hi, tile_d0
  g1[4] = (int)(rows & 0xffffu);                             // tile_d1, d2=0
  g1[5] = (int)stride_elems;                                 // d0_stride lo32
  g1[6] = 0;                                                 // strides hi
  g1[7] = 0;
  const v4i gz4 = {0, 0, 0, 0};
  const v8i gz8 = {0, 0, 0, 0, 0, 0, 0, 0};
  __builtin_amdgcn_tensor_load_to_lds(g0, g1, gz4, gz4, gz8, 0);
}

// ---------------------------------------------------------------------------
// Tiled batched GEMM: C = alpha * A[M,K] x B (+bias)
//  TA/TB float => convert to bf16 while staging (sync path, single buffer)
//  B_NK        => B stored [N,K] row-major (compute A*B^T)
//  OUT_TRANS   => C stored [N,M] bf16 (contiguous-M 16B stores per lane)
// Block tile 128x128, BK=32. 256 thr = 8 waves (4x2); wave tile 32x64 =
// 2x4 WMMA 16x16 accs => 8 v_wmma_f32_16x16x32_bf16 per k-step per wave.
// ---------------------------------------------------------------------------
template <typename TA, typename TB, bool B_NK, bool OUT_BF16, bool OUT_TRANS,
          bool HAS_BIAS>
__global__ __launch_bounds__(256) void gemm_wmma_kernel(
    const void* __restrict__ Ap, const void* __restrict__ Bp,
    const float* __restrict__ bias, void* __restrict__ Cp, int K, int lda,
    int ldb, int ldc, long long sA, long long sB, long long sC, float alpha) {
  constexpr bool ASYNC = (sizeof(TA) == 2) && (sizeof(TB) == 2) && B_NK;
  __shared__ SharedTiles sh;

  const int tid  = threadIdx.x;
  const int lane = tid & 31;
  const int wave = tid >> 5;
  const int wm   = wave & 3;   // 4 waves along M (32 rows each)
  const int wn   = wave >> 2;  // 2 waves along N (64 cols each)
  const int bm0  = blockIdx.y * BM;
  const int bn0  = blockIdx.x * BN;
  const int bz   = blockIdx.z;

  const TA* A = (const TA*)Ap + (size_t)bz * (size_t)sA;
  const TB* B = (const TB*)Bp + (size_t)bz * (size_t)sB;

  v8f acc[2][4];
#pragma unroll
  for (int i = 0; i < 2; ++i)
#pragma unroll
    for (int j = 0; j < 4; ++j)
#pragma unroll
      for (int r = 0; r < 8; ++r) acc[i][j][r] = 0.0f;

  // WMMA bf16 fragment lane mapping (ISA 7.12.2)
  const int mrow = lane & 15;
  const int half = lane >> 4;
  const int kbA  = half * 8;
  const int kbB  = half * 8;

  auto compute_tile = [&](int buf) {
    v16bf afr[2], bfr[4];
#pragma unroll
    for (int t = 0; t < 2; ++t) {
      const int row = wm * 32 + t * 16 + mrow;
      const v8bf lo = *(const v8bf*)&sh.As[buf][row][kbA];
      const v8bf hi = *(const v8bf*)&sh.As[buf][row][kbA + 16];
      afr[t] = __builtin_shufflevector(lo, hi, 0, 1, 2, 3, 4, 5, 6, 7, 8, 9, 10,
                                       11, 12, 13, 14, 15);
    }
#pragma unroll
    for (int t = 0; t < 4; ++t) {
      const int col = wn * 64 + t * 16 + mrow;
      const v8bf lo = *(const v8bf*)&sh.Bs[buf][col][kbB];
      const v8bf hi = *(const v8bf*)&sh.Bs[buf][col][kbB + 16];
      bfr[t] = __builtin_shufflevector(lo, hi, 0, 1, 2, 3, 4, 5, 6, 7, 8, 9, 10,
                                       11, 12, 13, 14, 15);
    }
#pragma unroll
    for (int tm = 0; tm < 2; ++tm)
#pragma unroll
      for (int tn = 0; tn < 4; ++tn)
        acc[tm][tn] = __builtin_amdgcn_wmma_f32_16x16x32_bf16(
            false, afr[tm], false, bfr[tn], (short)0, acc[tm][tn], false,
            false);
  };

  if constexpr (ASYNC) {
    // ---- double-buffered bf16 path: A via async-LDS, B via TDM ----
    const uint64_t gA = (uint64_t)A;
    const uint64_t gB = (uint64_t)B;
    auto stage = [&](int buf, int k0) {
      // A tile 128x32 bf16 = 512 x b128 async copies, 2 per thread
#pragma unroll
      for (int p = 0; p < 2; ++p) {
        const int idx = tid + p * 256;
        const int r = idx >> 2, c8 = (idx & 3) * 8;
        async_copy_b128(lds_off_A(buf, r, c8), gA,
                        (uint32_t)(((bm0 + r) * lda + (k0 + c8)) * 2));
      }
      // B tile 128x32 bf16: ONE tensor DMA descriptor, wave 0 issues
      if (wave == 0)
        tdm_load_tile_2d(lds_off_B(buf, 0, 0),
                         gB + (uint64_t)((bn0 * ldb + k0) * 2), (uint32_t)BN,
                         (uint32_t)ldb);
    };
    stage(0, 0);
    for (int k0 = 0; k0 < K; k0 += BK) {
      const int buf = (k0 >> 5) & 1;
      wait_async0();  // own A-tile async writes landed
      if (wave == 0) __builtin_amdgcn_s_wait_tensorcnt(0);  // B tile landed
      __syncthreads();  // publish to all waves
      if (k0 + BK < K) stage(buf ^ 1, k0 + BK);  // prefetch next tile
      compute_tile(buf);
      // no trailing barrier: ping-pong + top-of-loop barrier cover reuse
    }
  } else {
    // ---- sync fp32->bf16 staging (projection GEMMs) ----
    const float* Af = (const float*)A;
    const float* Bf = (const float*)B;
    for (int k0 = 0; k0 < K; k0 += BK) {
      {
        const int r0 = tid >> 3, c4 = (tid & 7) * 4;
#pragma unroll
        for (int p = 0; p < 4; ++p) {
          const int row = r0 + p * 32;
          const float4 v =
              *(const float4*)(Af + (size_t)(bm0 + row) * lda + (k0 + c4));
          v4bf o;
          o[0] = (bf16_t)v.x; o[1] = (bf16_t)v.y;
          o[2] = (bf16_t)v.z; o[3] = (bf16_t)v.w;
          *(v4bf*)&sh.As[0][row][c4] = o;
        }
      }
      {
#pragma unroll
        for (int p = 0; p < 4; ++p) {
          const int idx = tid + p * 256;
          const int kk = idx >> 5, n4 = (idx & 31) * 4;
          const float4 v =
              *(const float4*)(Bf + (size_t)(k0 + kk) * ldb + (bn0 + n4));
          sh.Bs[0][n4 + 0][kk] = (bf16_t)v.x;
          sh.Bs[0][n4 + 1][kk] = (bf16_t)v.y;
          sh.Bs[0][n4 + 2][kk] = (bf16_t)v.z;
          sh.Bs[0][n4 + 3][kk] = (bf16_t)v.w;
        }
      }
      __syncthreads();
      compute_tile(0);
      __syncthreads();
    }
  }

  // -------- epilogue (C layout per ISA 7.12.2) --------
#pragma unroll
  for (int tm = 0; tm < 2; ++tm) {
#pragma unroll
    for (int tn = 0; tn < 4; ++tn) {
      const int row0 = bm0 + wm * 32 + tm * 16 + half * 8;
      const int col  = bn0 + wn * 64 + tn * 16 + mrow;
      float bval = 0.0f;
      if constexpr (HAS_BIAS) bval = bias[col];
      if constexpr (OUT_TRANS) {
        // C[N,M] bf16: the 8 acc rows are M-consecutive -> single 16B store
        v8bf o;
#pragma unroll
        for (int r = 0; r < 8; ++r)
          o[r] = (bf16_t)(acc[tm][tn][r] * alpha + bval);
        bf16_t* C =
            (bf16_t*)Cp + (size_t)bz * (size_t)sC + (size_t)col * ldc + row0;
        *(v8bf*)C = o;
      } else {
        const size_t cbase =
            (size_t)bz * (size_t)sC + (size_t)row0 * ldc + col;
#pragma unroll
        for (int r = 0; r < 8; ++r) {
          const float v = acc[tm][tn][r] * alpha + bval;
          if constexpr (OUT_BF16)
            ((bf16_t*)Cp)[cbase + (size_t)r * ldc] = (bf16_t)v;
          else
            ((float*)Cp)[cbase + (size_t)r * ldc] = v;
        }
      }
    }
  }
}

// ---------------------------------------------------------------------------
// Row softmax, in-place fp32 -> bf16 (probs at same row base, pitch 2*Skv
// bf16). One block of 256 per row; wave32 shuffles + LDS cross-wave reduce.
// ---------------------------------------------------------------------------
__global__ __launch_bounds__(256) void softmax_rows_kernel(
    float* __restrict__ S, const float* __restrict__ mask, int Skv,
    int rows_per_batch) {
  const int row  = blockIdx.x;
  const int b    = row / rows_per_batch;
  const int tid  = threadIdx.x;
  const int lane = tid & 31;
  const int wave = tid >> 5;
  float* Srow       = S + (size_t)row * Skv;
  const float* mrow = mask + (size_t)b * Skv;
  const int h = Skv / 2;  // 1024

  float v[8];
  {
    const float4 a  = *(const float4*)(Srow + 4 * tid);
    const float4 c  = *(const float4*)(Srow + h + 4 * tid);
    const float4 ma = *(const float4*)(mrow + 4 * tid);
    const float4 mc = *(const float4*)(mrow + h + 4 * tid);
    v[0] = a.x + (1.0f - ma.x) * -10000.0f;
    v[1] = a.y + (1.0f - ma.y) * -10000.0f;
    v[2] = a.z + (1.0f - ma.z) * -10000.0f;
    v[3] = a.w + (1.0f - ma.w) * -10000.0f;
    v[4] = c.x + (1.0f - mc.x) * -10000.0f;
    v[5] = c.y + (1.0f - mc.y) * -10000.0f;
    v[6] = c.z + (1.0f - mc.z) * -10000.0f;
    v[7] = c.w + (1.0f - mc.w) * -10000.0f;
  }

  __shared__ float red[8];

  float mx = v[0];
#pragma unroll
  for (int j = 1; j < 8; ++j) mx = fmaxf(mx, v[j]);
  for (int o = 16; o > 0; o >>= 1) mx = fmaxf(mx, __shfl_xor(mx, o, 32));
  if (lane == 0) red[wave] = mx;
  __syncthreads();
  if (wave == 0) {
    float t = (lane < 8) ? red[lane] : -3.0e38f;
    for (int o = 4; o > 0; o >>= 1) t = fmaxf(t, __shfl_xor(t, o, 32));
    if (lane == 0) red[0] = t;
  }
  __syncthreads();
  mx = red[0];

  float s = 0.0f;
#pragma unroll
  for (int j = 0; j < 8; ++j) {
    v[j] = __expf(v[j] - mx);
    s += v[j];
  }
  for (int o = 16; o > 0; o >>= 1) s += __shfl_xor(s, o, 32);
  __syncthreads();
  if (lane == 0) red[wave] = s;
  __syncthreads();
  if (wave == 0) {
    float t = (lane < 8) ? red[lane] : 0.0f;
    for (int o = 4; o > 0; o >>= 1) t += __shfl_xor(t, o, 32);
    if (lane == 0) red[0] = t;
  }
  __syncthreads();
  const float inv = 1.0f / red[0];

  bf16_t* P = (bf16_t*)Srow;  // in-place
  v4bf o0, o1;
#pragma unroll
  for (int j = 0; j < 4; ++j) {
    o0[j] = (bf16_t)(v[j] * inv);
    o1[j] = (bf16_t)(v[4 + j] * inv);
  }
  *(v4bf*)(P + 4 * tid)     = o0;
  *(v4bf*)(P + h + 4 * tid) = o1;
}

// ---------------------------------------------------------------------------
// workspace (224 MiB):
//   Qb bf16 [B*Sq*H] row-major     33.6 MB
//   Kb bf16 [B*Skv*H] row-major    33.6 MB
//   Vt bf16 [B][H][Skv] TRANSPOSED 33.6 MB
//   S  fp32 [B*Sq*Skv]            134.2 MB (reused in-place for bf16 P)
// ---------------------------------------------------------------------------
extern "C" void kernel_launch(void* const* d_in, const int* in_sizes, int n_in,
                              void* d_out, int out_size, void* d_ws,
                              size_t ws_size, hipStream_t stream) {
  (void)in_sizes; (void)n_in; (void)out_size; (void)ws_size;
  const int B = 8, Sq = 2048, Skv = 2048, H = 1024;

  const float* q   = (const float*)d_in[0];
  const float* k   = (const float*)d_in[1];
  const float* msk = (const float*)d_in[2];
  const float* Wq  = (const float*)d_in[3];
  const float* bq  = (const float*)d_in[4];
  const float* Wk  = (const float*)d_in[5];
  const float* bk  = (const float*)d_in[6];
  const float* Wv  = (const float*)d_in[7];
  const float* bv  = (const float*)d_in[8];

  bf16_t* Qb = (bf16_t*)d_ws;
  bf16_t* Kb = Qb + (size_t)B * Sq * H;
  bf16_t* Vt = Kb + (size_t)B * Skv * H;
  float*  Sc = (float*)(Vt + (size_t)B * Skv * H);

  const dim3 blk(256);

  // Q, K projections: [16384,1024] x [1024,1024] (fp32 -> bf16, row-major)
  const dim3 gqk(H / BN, (B * Sq) / BM, 1);
  gemm_wmma_kernel<float, float, false, true, false, true>
      <<<gqk, blk, 0, stream>>>(q, Wq, bq, Qb, H, H, H, H, 0, 0, 0, 1.0f);
  gemm_wmma_kernel<float, float, false, true, false, true>
      <<<gqk, blk, 0, stream>>>(k, Wk, bk, Kb, H, H, H, H, 0, 0, 0, 1.0f);

  // V projection with transposed output: Vt[b][d][s]
  const dim3 gv(H / BN, Sq / BM, B);
  gemm_wmma_kernel<float, float, false, true, true, true>
      <<<gv, blk, 0, stream>>>(k, Wv, bv, Vt, H, H, H, /*ldc(T)=*/Sq,
                               (long long)Sq * H, 0, (long long)H * Sq, 1.0f);

  // scores: S[b] = (1/sqrt(H)) * Q[b] x K[b]^T  (async A + TDM B, dbl-buffer)
  const dim3 gs(Skv / BN, Sq / BM, B);
  gemm_wmma_kernel<bf16_t, bf16_t, true, false, false, false>
      <<<gs, blk, 0, stream>>>(Qb, Kb, nullptr, Sc, H, H, H, Skv,
                               (long long)Sq * H, (long long)Skv * H,
                               (long long)Sq * Skv, 0.03125f);

  // softmax rows, in-place
  softmax_rows_kernel<<<dim3(B * Sq), blk, 0, stream>>>(Sc, msk, Skv, Sq);

  // O[b] = P[b] x Vt[b]^T -> d_out fp32
  const dim3 gpv(H / BN, Sq / BM, B);
  gemm_wmma_kernel<bf16_t, bf16_t, true, false, false, false>
      <<<gpv, blk, 0, stream>>>((const bf16_t*)Sc, Vt, nullptr, (float*)d_out,
                                Skv, 2 * Skv, Skv, H,
                                (long long)Sq * 2 * Skv, (long long)Skv * H,
                                (long long)Sq * H, 1.0f);
}
